// CausalSelfAttention_64871186039064
// MI455X (gfx1250) — compile-verified, matrix operand order checked by reading, not run on hardware
//
#include <hip/hip_runtime.h>
#include <hip/hip_bf16.h>
#include <stdint.h>

// CDNA5 / gfx1250, wave32. Matrix math via v_wmma_f32_16x16x32_f16; tile
// staging via GLOBAL_LOAD_ASYNC_TO_LDS_B128 (ASYNCcnt) per ISA 15.18.3;
// softmax row reductions via ds_swizzle_b32 XOR patterns (ISA 11.2).
typedef __attribute__((ext_vector_type(16))) _Float16 v16h;
typedef __attribute__((ext_vector_type(8)))  float    v8f;

#define TSEQ   2048
#define DMODEL 2048
#define NH     16
#define HD     128
#define LDQKV  6144   // 3*DMODEL: [q | k | v] each [NH][HD]

// ---------------------------------------------------------------------------
// Async global->LDS 16-byte copy (no VGPR round trip, tracked by ASYNCcnt).
// Generic pointers to __shared__ are {aperture_hi, lds_offset}; the low 32
// bits are exactly the VDST LDS byte address the instruction consumes.
// ---------------------------------------------------------------------------
__device__ __forceinline__ void async_copy_b128(const _Float16* gsrc,
                                                const _Float16* ldst) {
  unsigned lds_off = (unsigned)(uintptr_t)ldst;
  unsigned long long gaddr = (unsigned long long)(uintptr_t)gsrc;
  asm volatile("global_load_async_to_lds_b128 %0, %1, off"
               :
               : "v"(lds_off), "v"(gaddr)
               : "memory");
}
__device__ __forceinline__ void wait_async0() {
  asm volatile("s_wait_asynccnt 0x0" ::: "memory");
}
__device__ __forceinline__ void wait_async6() {  // allow 1 prefetch stage (6 ops) in flight
  asm volatile("s_wait_asynccnt 0x6" ::: "memory");
}

// XOR-lane exchange via ds_swizzle_b32 (group-of-32: and=0x1f, xor=MASK<<10)
template <int XORMASK>
__device__ __forceinline__ float swz_xor(float v) {
  int r = __builtin_amdgcn_ds_swizzle(__float_as_int(v), 0x001f | (XORMASK << 10));
  return __int_as_float(r);
}
// Reduce across the 16-lane half (rows of one 16x16 C/D accumulator half)
__device__ __forceinline__ float rowred_max(float v) {
  v = fmaxf(v, swz_xor<1>(v));
  v = fmaxf(v, swz_xor<2>(v));
  v = fmaxf(v, swz_xor<4>(v));
  v = fmaxf(v, swz_xor<8>(v));
  return v;
}
__device__ __forceinline__ float rowred_sum(float v) {
  v += swz_xor<1>(v);
  v += swz_xor<2>(v);
  v += swz_xor<4>(v);
  v += swz_xor<8>(v);
  return v;
}

// ---------------------------------------------------------------------------
// Fragment loaders for the CDNA5 WMMA 16x16x32 f16 layouts (ISA 7.12.2).
// A (16x32, MxK), per lane: row = lane&15; lanes 0-15 hold K=[0..7]+[16..23],
// lanes 16-31 hold K=[8..15]+[24..31]; 2x 16B contiguous chunks per lane.
// B (32x16, KxN) uses the same packing with "row" = output column N, i.e. a
// B-fragment is an A-style load of B^T — so B operands are kept N-major.
// ---------------------------------------------------------------------------
__device__ __forceinline__ v16h load_frag16x32(const _Float16* base, int ld, int lane) {
  const int row = lane & 15;
  const int hi  = (lane >> 4) & 1;
  const _Float16* p0 = base + (size_t)row * ld + hi * 8;
  const _Float16* p1 = p0 + 16;
  v16h f;
#pragma unroll
  for (int i = 0; i < 8; ++i) { f[i] = p0[i]; f[8 + i] = p1[i]; }
  return f;
}

__device__ __forceinline__ v8f wmma_f16(v16h a, v16h b, v8f c) {
  return __builtin_amdgcn_wmma_f32_16x16x32_f16(false, a, false, b, (short)0, c,
                                                false, false);
}

// ---------------------------------------------------------------------------
// Elementwise fp32 -> f16 convert
// ---------------------------------------------------------------------------
__global__ __launch_bounds__(256) void cvt_f32_to_f16_kernel(
    const float* __restrict__ in, _Float16* __restrict__ out, int n) {
  int i = blockIdx.x * 256 + threadIdx.x;
  if (i < n) out[i] = (_Float16)in[i];
}

// in[K][N] fp32 -> out[N][K] f16 (transpose so GEMM B-fragments are contiguous)
__global__ __launch_bounds__(256) void transpose_cvt_kernel(
    const float* __restrict__ in, _Float16* __restrict__ out, int K, int N) {
  int idx = blockIdx.x * 256 + threadIdx.x;
  if (idx < K * N) {
    int k = idx / N;
    int n = idx - k * N;
    out[(size_t)n * K + k] = (_Float16)in[idx];
  }
}

// ---------------------------------------------------------------------------
// GEMM: C[M,N] = A[M,K] * B[K,N], A f16 row-major, Bt = B^T f16 [N,K].
// Block = 256 threads = 8 waves; block tile 64(M) x 128(N); K-step 64.
// Double-buffered LDS staging with async global->LDS b128 copies:
//   A tile 64x64 (2 chunks/thread) + B tile 128x64 (4 chunks/thread).
// Prefetch next K-step; s_wait_asynccnt 6 keeps it in flight while the
// ready buffer feeds ds_load_b128 fragments -> 8 WMMAs per wave per stage.
// LDS: 2*(8KB + 16KB) = 48KB of the 320KB WGP pool.
// ---------------------------------------------------------------------------
template <int OUT_F16>
__global__ __launch_bounds__(256) void gemm_wmma_kernel(
    const _Float16* __restrict__ A, const _Float16* __restrict__ Bt,
    void* __restrict__ Cout, int M, int N, int K) {
  __shared__ _Float16 Abuf[2][64 * 64];    // 16 KB
  __shared__ _Float16 Bbuf[2][128 * 64];   // 32 KB
  const int tid  = threadIdx.x;
  const int lane = tid & 31;
  const int wave = tid >> 5;
  const int mBlk = blockIdx.y * 64;
  const int nBlk = blockIdx.x * 128;
  const int m0w  = (wave >> 2) * 32;   // wave tile origin within block tile
  const int n0w  = (wave & 3) * 32;

  // staging coordinates (16B chunks: 8 halves); rows of 64 halves = 8 chunks
  const int srow = tid >> 3;           // 0..31
  const int sch  = (tid & 7) * 8;      // 0..56

  v8f acc00 = {}, acc01 = {}, acc10 = {}, acc11 = {};

#define GEMM_STAGE(buf, kk)                                                        \
  do {                                                                             \
    _Pragma("unroll")                                                              \
    for (int t = 0; t < 2; ++t) /* A: 64 rows */                                   \
      async_copy_b128(A + (size_t)(mBlk + srow + 32 * t) * K + (kk) + sch,         \
                      &Abuf[buf][(srow + 32 * t) * 64 + sch]);                     \
    _Pragma("unroll")                                                              \
    for (int t = 0; t < 4; ++t) /* B: 128 rows */                                  \
      async_copy_b128(Bt + (size_t)(nBlk + srow + 32 * t) * K + (kk) + sch,        \
                      &Bbuf[buf][(srow + 32 * t) * 64 + sch]);                     \
  } while (0)

  GEMM_STAGE(0, 0);
  for (int k = 0; k < K; k += 64) {
    const int buf = (k >> 6) & 1;
    const bool more = (k + 64) < K;
    if (more) GEMM_STAGE(buf ^ 1, k + 64);   // prefetch next stage
    if (more) wait_async6(); else wait_async0();
    __syncthreads();                         // all waves' slices visible

#pragma unroll
    for (int kc = 0; kc < 64; kc += 32) {
      v16h a0 = load_frag16x32(&Abuf[buf][(size_t)m0w * 64 + kc], 64, lane);
      v16h a1 = load_frag16x32(&Abuf[buf][(size_t)(m0w + 16) * 64 + kc], 64, lane);
      v16h b0 = load_frag16x32(&Bbuf[buf][(size_t)n0w * 64 + kc], 64, lane);
      v16h b1 = load_frag16x32(&Bbuf[buf][(size_t)(n0w + 16) * 64 + kc], 64, lane);
      acc00 = wmma_f16(a0, b0, acc00);
      acc01 = wmma_f16(a0, b1, acc01);
      acc10 = wmma_f16(a1, b0, acc10);
      acc11 = wmma_f16(a1, b1, acc11);
    }
    __syncthreads();                         // done reading before re-stage
  }
#undef GEMM_STAGE

  // C/D layout: lane&15 = N, (lane>>4)*8 + reg = M (ISA 7.12.2)
  const int nn = lane & 15;
  const int mb = (lane >> 4) * 8;
  const int m0 = mBlk + m0w;
  const int n0 = nBlk + n0w;
#pragma unroll
  for (int r = 0; r < 8; ++r) {
    size_t row0 = (size_t)(m0 + mb + r) * N;
    size_t row1 = (size_t)(m0 + 16 + mb + r) * N;
    if (OUT_F16) {
      _Float16* C = (_Float16*)Cout;
      C[row0 + n0 + nn]      = (_Float16)acc00[r];
      C[row0 + n0 + 16 + nn] = (_Float16)acc01[r];
      C[row1 + n0 + nn]      = (_Float16)acc10[r];
      C[row1 + n0 + 16 + nn] = (_Float16)acc11[r];
    } else {
      float* C = (float*)Cout;
      C[row0 + n0 + nn]      = acc00[r];
      C[row0 + n0 + 16 + nn] = acc01[r];
      C[row1 + n0 + nn]      = acc10[r];
      C[row1 + n0 + 16 + nn] = acc11[r];
    }
  }
}

// ---------------------------------------------------------------------------
// Flash attention (causal), one (b,h,qtile) per block. 256 threads = 8 waves,
// each wave owns 16 q-rows. kv tiles of 32. The shared K tile (32x128 f16) is
// async-copied to LDS once per block; V is staged transposed via ds stores so
// its B-fragments are kv-contiguous. Row m of a 16x16 f32 accumulator lives
// in reg (m&7) across one 16-lane half, so ds_swizzle XOR reductions yield
// row stats already laid out to match the accumulator -> elementwise rescale.
// ---------------------------------------------------------------------------
__global__ __launch_bounds__(256) void flash_attn_kernel(
    const _Float16* __restrict__ qkv, _Float16* __restrict__ attn) {
  __shared__ _Float16 Kbuf[32 * HD];      // [kv][d]  8 KB
  __shared__ _Float16 Vt[HD * 32];        // [d][kv]  8 KB
  __shared__ _Float16 Pbuf[8][16 * 32];   // per-wave P tile, 8 KB

  const int tid   = threadIdx.x;
  const int lane  = tid & 31;
  const int wave  = tid >> 5;
  const int bh    = blockIdx.y;
  const int b     = bh >> 4;
  const int h     = bh & 15;
  const int qbase = blockIdx.x * 128;
  const int qr    = qbase + wave * 16;
  const size_t rowbase = (size_t)b * TSEQ;

  // Q fragments: 16 rows x 128 d = 4 A-fragments, kept in VGPRs
  const _Float16* qp = qkv + (rowbase + qr) * LDQKV + h * HD;
  v16h qfrag[4];
#pragma unroll
  for (int kk = 0; kk < 4; ++kk)
    qfrag[kk] = load_frag16x32(qp + kk * 32, LDQKV, lane);

  v8f o[8];
#pragma unroll
  for (int dt = 0; dt < 8; ++dt) o[dt] = (v8f){};
  float mrow[8], lrow[8];
#pragma unroll
  for (int r = 0; r < 8; ++r) { mrow[r] = -1e30f; lrow[r] = 0.0f; }

  const float scale = 0.08838834764831845f;  // 1/sqrt(128)
  const int mb = (lane >> 4) * 8;
  const int nn = lane & 15;

  // K-tile staging coords: 32 rows x 128 halves = 512 16B-chunks, 2/thread
  const int krow0 = tid >> 4;              // 0..15
  const int kch   = (tid & 15) * 8;        // 0..120

  for (int kv0 = 0; kv0 < qbase + 128; kv0 += 32) {
    // Async-stage K tile [kv][d] into LDS (shared by all 8 waves)
    {
      const _Float16* kg = qkv + (rowbase + kv0) * LDQKV + DMODEL + h * HD;
      async_copy_b128(kg + (size_t)krow0 * LDQKV + kch, &Kbuf[krow0 * HD + kch]);
      async_copy_b128(kg + (size_t)(krow0 + 16) * LDQKV + kch,
                      &Kbuf[(krow0 + 16) * HD + kch]);
    }
    // Stage V^T tile: Vt[d][kv] (transposed scatter, coalesced read along d)
    {
      const int kv = tid >> 3;              // 0..31
      const int d0 = (tid & 7) * 16;        // 0..112
      const _Float16* vp =
          qkv + (rowbase + kv0 + kv) * LDQKV + 2 * DMODEL + h * HD + d0;
#pragma unroll
      for (int j = 0; j < 16; ++j) Vt[(d0 + j) * 32 + kv] = vp[j];
    }
    wait_async0();
    __syncthreads();

    if (kv0 <= qr + 15) {  // wave has unmasked work in this kv tile
      // S = Q K^T : two 16x16 tiles (kv cols 0-15, 16-31), 4 d-chunks each
      v8f s0 = {}, s1 = {};
#pragma unroll
      for (int kk = 0; kk < 4; ++kk) {
        v16h bk0 = load_frag16x32(&Kbuf[kk * 32], HD, lane);
        v16h bk1 = load_frag16x32(&Kbuf[16 * HD + kk * 32], HD, lane);
        s0 = wmma_f16(qfrag[kk], bk0, s0);
        s1 = wmma_f16(qfrag[kk], bk1, s1);
      }

      // scale + causal mask + online softmax update
      float p0[8], p1[8];
#pragma unroll
      for (int r = 0; r < 8; ++r) {
        const int qg = qr + mb + r;
        float v0 = s0[r] * scale;
        float v1 = s1[r] * scale;
        if (kv0 + nn > qg)      v0 = -1e30f;
        if (kv0 + 16 + nn > qg) v1 = -1e30f;
        const float mx    = rowred_max(fmaxf(v0, v1));   // 16-lane row reduce
        const float mnew  = fmaxf(mrow[r], mx);
        const float alpha = __expf(mrow[r] - mnew);
        const float e0 = __expf(v0 - mnew);
        const float e1 = __expf(v1 - mnew);
        const float sum = rowred_sum(e0 + e1);
        lrow[r] = lrow[r] * alpha + sum;
        mrow[r] = mnew;
        p0[r] = e0;
        p1[r] = e1;
#pragma unroll
        for (int dt = 0; dt < 8; ++dt) o[dt][r] *= alpha;
      }

      // P: accumulator layout -> LDS -> A-fragment layout (f16)
#pragma unroll
      for (int r = 0; r < 8; ++r) {
        Pbuf[wave][(mb + r) * 32 + nn]      = (_Float16)p0[r];
        Pbuf[wave][(mb + r) * 32 + 16 + nn] = (_Float16)p1[r];
      }
      v16h pfrag = load_frag16x32(&Pbuf[wave][0], 32, lane);

      // O += P @ V : 8 d-tiles of 16, B-fragments from transposed V in LDS
#pragma unroll
      for (int dt = 0; dt < 8; ++dt) {
        v16h vfrag = load_frag16x32(&Vt[dt * 16 * 32], 32, lane);
        o[dt] = wmma_f16(pfrag, vfrag, o[dt]);
      }
    }
    __syncthreads();  // protect Kbuf/Vt before next tile overwrites
  }

  // Normalize and write attn output in [B,T,H*Hd] f16 layout for final GEMM
#pragma unroll
  for (int r = 0; r < 8; ++r) {
    const float inv = 1.0f / lrow[r];
    const int q = qr + mb + r;
    _Float16* op = attn + ((size_t)(b * TSEQ + q)) * DMODEL + h * HD;
#pragma unroll
    for (int dt = 0; dt < 8; ++dt)
      op[dt * 16 + nn] = (_Float16)(o[dt][r] * inv);
  }
}

// ---------------------------------------------------------------------------
// Host-side orchestration (graph-capture safe: all launches on `stream`)
// ---------------------------------------------------------------------------
extern "C" void kernel_launch(void* const* d_in, const int* in_sizes, int n_in,
                              void* d_out, int out_size, void* d_ws, size_t ws_size,
                              hipStream_t stream) {
  (void)in_sizes; (void)n_in; (void)out_size; (void)ws_size;
  const float* x    = (const float*)d_in[0];  // [2,2048,2048]
  const float* Wqkv = (const float*)d_in[1];  // [2048,6144]
  const float* Wout = (const float*)d_in[2];  // [2048,2048]

  const size_t N_X    = (size_t)2 * TSEQ * DMODEL;   //  8,388,608
  const size_t N_WQKV = (size_t)DMODEL * LDQKV;      // 12,582,912
  const size_t N_WOUT = (size_t)DMODEL * DMODEL;     //  4,194,304
  const size_t N_QKV  = (size_t)2 * TSEQ * LDQKV;    // 25,165,824

  _Float16* ws     = (_Float16*)d_ws;
  _Float16* x16    = ws;                  // f16 activations
  _Float16* wqkvT  = x16 + N_X;           // W_qkv^T f16 [6144,2048]
  _Float16* woutT  = wqkvT + N_WQKV;      // W_out^T f16 [2048,2048]
  _Float16* qkv    = woutT + N_WOUT;      // [2,2048,6144] f16
  _Float16* attnws = qkv + N_QKV;         // [2,2048,2048] f16

  // 1) precision conversion (+ transpose of weights for B-fragment loads)
  cvt_f32_to_f16_kernel<<<(int)((N_X + 255) / 256), 256, 0, stream>>>(x, x16, (int)N_X);
  transpose_cvt_kernel<<<(int)((N_WQKV + 255) / 256), 256, 0, stream>>>(Wqkv, wqkvT, DMODEL, LDQKV);
  transpose_cvt_kernel<<<(int)((N_WOUT + 255) / 256), 256, 0, stream>>>(Wout, woutT, DMODEL, DMODEL);

  // 2) QKV projection: [4096,2048] x [2048,6144] -> f16 qkv
  gemm_wmma_kernel<1><<<dim3(LDQKV / 128, (2 * TSEQ) / 64), 256, 0, stream>>>(
      x16, wqkvT, (void*)qkv, 2 * TSEQ, LDQKV, DMODEL);

  // 3) causal flash attention per (b,h): grid (qtiles=16, b*h=32)
  flash_attn_kernel<<<dim3(TSEQ / 128, 2 * NH), 256, 0, stream>>>(qkv, attnws);

  // 4) output projection: [4096,2048] x [2048,2048] -> fp32 d_out
  gemm_wmma_kernel<0><<<dim3(DMODEL / 128, (2 * TSEQ) / 64), 256, 0, stream>>>(
      attnws, woutT, d_out, 2 * TSEQ, DMODEL, DMODEL);
}
